// FeatureCorrelation_80960133530196
// MI455X (gfx1250) — compile-verified, matrix operand order checked by reading, not run on hardware
//
#include <hip/hip_runtime.h>
#include <hip/hip_bf16.h>

typedef __attribute__((ext_vector_type(16))) __bf16 v16bf;
typedef __attribute__((ext_vector_type(8)))  float  v8f;
typedef __attribute__((ext_vector_type(4)))  int    v4i;

typedef __attribute__((address_space(1))) v4i* gptr4;
typedef __attribute__((address_space(3))) v4i* lptr4;

#define NB   16
#define CCH  512
#define HWW  48
#define HW   2304   // 48*48

#if __has_builtin(__builtin_amdgcn_global_load_async_to_lds_b128) && \
    __has_builtin(__builtin_amdgcn_s_wait_asynccnt)
#define USE_ASYNC_LDS 1
#else
#define USE_ASYNC_LDS 0
#endif

// ---------------------------------------------------------------------------
// Kernel 1: convert f32 -> bf16 and transpose both features into [b][l][c]
//   At[b][l][c] = A[b, c, l%48, l/48]   (w-major flatten of A)
//   Bt[b][k][c] = B[b, c, k/48, k%48]   (h-major flatten of B)
// Writes are fully coalesced over c; strided reads ride the 192MB L2.
// ---------------------------------------------------------------------------
__global__ void pack_kernel(const float* __restrict__ A, const float* __restrict__ B,
                            __bf16* __restrict__ At, __bf16* __restrict__ Bt) {
    size_t t = (size_t)blockIdx.x * blockDim.x + threadIdx.x;
    const size_t total = (size_t)NB * HW * CCH;
    if (t >= total) return;
    int c = (int)(t % CCH);
    int l = (int)((t / CCH) % HW);
    int b = (int)(t / ((size_t)CCH * HW));
    size_t srcBase = ((size_t)b * CCH + c) * HW;
    float av = A[srcBase + (size_t)(l % HWW) * HWW + (size_t)(l / HWW)];
    float bv = B[srcBase + l];
    At[t] = (__bf16)av;
    Bt[t] = (__bf16)bv;
}

// ---------------------------------------------------------------------------
// Kernel 2: per-batch GEMM  D[l,k] = sum_c At[l,c]*Bt[k,c], ReLU, store,
// and atomically accumulate sum_l relu(v)^2 into normsq[b][k].
// Block: 256 threads (8 waves), tile 128(M) x 128(N), K-step 32 (bf16 WMMA).
// Double-buffered LDS (2 x 20KB); async global->LDS copies overlap the WMMAs
// of the current K-tile with the fetch of the next one (ASYNCcnt pipeline).
// ---------------------------------------------------------------------------
__global__ __launch_bounds__(256)
void gemm_relu_kernel(const __bf16* __restrict__ At, const __bf16* __restrict__ Bt,
                      float* __restrict__ out, float* __restrict__ normsq) {
    __shared__ uint4 lds[2560];  // buf*1280; within buf: [0,640)=A, [640,1280)=B

    const int tid   = threadIdx.x;
    const int b     = blockIdx.z;
    const int m0    = blockIdx.y * 128;   // l-tile origin
    const int n0    = blockIdx.x * 128;   // k-tile origin

    const int lane   = tid & 31;
    const int laneLo = lane & 15;
    const int laneHi = lane >> 4;
    const int wave   = tid >> 5;
    const int waveM  = wave >> 2;         // 0..1 -> 64 rows each
    const int waveN  = wave & 3;          // 0..3 -> 32 cols each

    const __bf16* gA = At + ((size_t)b * HW + m0) * CCH;
    const __bf16* gB = Bt + ((size_t)b * HW + n0) * CCH;

    // stage one 128x32 K-tile of A and B into LDS buffer `buf`
    auto stage = [&](int buf, int kk) {
#pragma unroll
        for (int i = 0; i < 2; ++i) {
            int idx = i * 256 + tid;
            int row = idx >> 2;
            int seg = idx & 3;
            size_t go = (size_t)row * CCH + kk + seg * 8;
            int lo = buf * 1280 + row * 5 + seg;
#if USE_ASYNC_LDS
            __builtin_amdgcn_global_load_async_to_lds_b128(
                (gptr4)(gA + go), (lptr4)&lds[lo], 0, 0);
            __builtin_amdgcn_global_load_async_to_lds_b128(
                (gptr4)(gB + go), (lptr4)&lds[lo + 640], 0, 0);
#else
            uint4 va = *(const uint4*)(gA + go);
            uint4 vb = *(const uint4*)(gB + go);
            lds[lo]       = va;
            lds[lo + 640] = vb;
#endif
        }
    };

    v8f acc[4][2];
#pragma unroll
    for (int tm = 0; tm < 4; ++tm)
#pragma unroll
        for (int tn = 0; tn < 2; ++tn)
            acc[tm][tn] = 0.f;

    stage(0, 0);

    int buf = 0;
    for (int kk = 0; kk < CCH; kk += 32, buf ^= 1) {
#if USE_ASYNC_LDS
        __builtin_amdgcn_s_wait_asynccnt(0);
#endif
        __syncthreads();

        if (kk + 32 < CCH) {
            stage(buf ^ 1, kk + 32);
#if !USE_ASYNC_LDS
            __builtin_prefetch(gA + (size_t)(tid >> 1) * CCH + kk + 32, 0, 1);
            __builtin_prefetch(gB + (size_t)(tid >> 1) * CCH + kk + 32, 0, 1);
#endif
        }

        // ---- fragment loads: lane<16 takes K0-7 & K16-23, lane>=16 K8-15 & K24-31
        union FragU { uint4 u[2]; v16bf v; };
        const int base = buf * 1280;
        v16bf aF[4], bF[2];
#pragma unroll
        for (int tm = 0; tm < 4; ++tm) {
            int r = waveM * 64 + tm * 16 + laneLo;
            FragU f;
            f.u[0] = lds[base + r * 5 + laneHi];
            f.u[1] = lds[base + r * 5 + 2 + laneHi];
            aF[tm] = f.v;
        }
#pragma unroll
        for (int tn = 0; tn < 2; ++tn) {
            int r = waveN * 32 + tn * 16 + laneLo;
            FragU f;
            f.u[0] = lds[base + 640 + r * 5 + laneHi];
            f.u[1] = lds[base + 640 + r * 5 + 2 + laneHi];
            bF[tn] = f.v;
        }

        // ---- 8 WMMAs per wave per K-step
#pragma unroll
        for (int tm = 0; tm < 4; ++tm)
#pragma unroll
            for (int tn = 0; tn < 2; ++tn)
                acc[tm][tn] = __builtin_amdgcn_wmma_f32_16x16x32_bf16(
                    false, aF[tm], false, bF[tn], (short)0, acc[tm][tn], false, false);
    }

    // ---- epilogue: ReLU, store, per-column sum of squares
#pragma unroll
    for (int tn = 0; tn < 2; ++tn) {
        int n = n0 + waveN * 32 + tn * 16 + laneLo;
        float pn = 0.f;
#pragma unroll
        for (int tm = 0; tm < 4; ++tm) {
            int mBase = m0 + waveM * 64 + tm * 16 + laneHi * 8;
#pragma unroll
            for (int r = 0; r < 8; ++r) {
                float v = acc[tm][tn][r];
                v = v > 0.f ? v : 0.f;
                out[((size_t)b * HW + (mBase + r)) * HW + n] = v;
                pn += v * v;
            }
        }
        atomicAdd(&normsq[b * HW + n], pn);
    }
}

// ---------------------------------------------------------------------------
// Kernel 3: normsq -> rsqrt(normsq + eps) in place (36,864 elements)
// ---------------------------------------------------------------------------
__global__ void rnorm_kernel(float* __restrict__ ns, int n) {
    int i = blockIdx.x * blockDim.x + threadIdx.x;
    if (i < n) ns[i] = rsqrtf(ns[i] + 1e-6f);
}

// ---------------------------------------------------------------------------
// Kernel 4: out[b][l][k] *= rn[b][k], float4 vectorized (k fastest, 2304%4==0)
// ---------------------------------------------------------------------------
__global__ void scale_kernel(float* __restrict__ out, const float* __restrict__ rn) {
    size_t i = (size_t)blockIdx.x * blockDim.x + threadIdx.x;   // float4 index
    const size_t total4 = (size_t)NB * HW * HW / 4;
    if (i >= total4) return;
    size_t flat = i * 4;
    int b = (int)(flat / ((size_t)HW * HW));
    int k = (int)(flat % HW);
    float4 o = ((const float4*)out)[i];
    float4 r = *(const float4*)(rn + (size_t)b * HW + k);
    o.x *= r.x; o.y *= r.y; o.z *= r.z; o.w *= r.w;
    ((float4*)out)[i] = o;
}

// ---------------------------------------------------------------------------
extern "C" void kernel_launch(void* const* d_in, const int* in_sizes, int n_in,
                              void* d_out, int out_size, void* d_ws, size_t ws_size,
                              hipStream_t stream) {
    const float* A = (const float*)d_in[0];
    const float* B = (const float*)d_in[1];
    float* out = (float*)d_out;

    const size_t packElems = (size_t)NB * HW * CCH;            // 18,874,368
    __bf16* At = (__bf16*)d_ws;
    __bf16* Bt = At + packElems;
    float*  normsq = (float*)((char*)d_ws + 2 * packElems * sizeof(__bf16));

    (void)hipMemsetAsync(normsq, 0, (size_t)NB * HW * sizeof(float), stream);

    {
        unsigned g = (unsigned)((packElems + 255) / 256);
        pack_kernel<<<g, 256, 0, stream>>>(A, B, At, Bt);
    }
    {
        dim3 g(HW / 128, HW / 128, NB);                        // 18 x 18 x 16
        gemm_relu_kernel<<<g, 256, 0, stream>>>(At, Bt, out, normsq);
    }
    {
        int n = NB * HW;
        rnorm_kernel<<<(n + 255) / 256, 256, 0, stream>>>(normsq, n);
    }
    {
        size_t t4 = (size_t)NB * HW * HW / 4;
        scale_kernel<<<(unsigned)((t4 + 255) / 256), 256, 0, stream>>>(out, normsq);
    }
}
